// DisMax2LossFirstPart_75436805587334
// MI455X (gfx1250) — compile-verified, matrix operand order checked by reading, not run on hardware
//
#include <hip/hip_runtime.h>
#include <hip/hip_bf16.h>

typedef __attribute__((ext_vector_type(16))) __bf16 v16bf;
typedef __attribute__((ext_vector_type(8)))  float  v8f;

#define DIMK   512
#define NROWS  8192
#define NCOLS  10000

// ---------------------------------------------------------------------------
// Kernel 1: L2-normalize each row and split fp32 -> bf16 hi + bf16 lo planes.
// One wave32 per row; lane handles 16 contiguous floats.
// ---------------------------------------------------------------------------
__global__ __launch_bounds__(256) void norm_split_kernel(
    const float* __restrict__ feats, const float* __restrict__ protos,
    __bf16* __restrict__ fn_hi, __bf16* __restrict__ fn_lo,
    __bf16* __restrict__ pn_hi, __bf16* __restrict__ pn_lo)
{
  const int wave = blockIdx.x * (blockDim.x >> 5) + (threadIdx.x >> 5);
  const int lane = threadIdx.x & 31;
  const int NTOT = NROWS + NCOLS;
  if (wave >= NTOT) return;

  const float* src;
  __bf16 *dhi, *dlo;
  if (wave < NROWS) {
    src = feats  + (size_t)wave * DIMK;
    dhi = fn_hi  + (size_t)wave * DIMK;
    dlo = fn_lo  + (size_t)wave * DIMK;
  } else {
    const int r = wave - NROWS;
    src = protos + (size_t)r * DIMK;
    dhi = pn_hi  + (size_t)r * DIMK;
    dlo = pn_lo  + (size_t)r * DIMK;
  }

  float4 x[4];
  const float4* s4 = (const float4*)(src + lane * 16);
  float ss = 0.f;
#pragma unroll
  for (int i = 0; i < 4; ++i) {
    x[i] = s4[i];
    ss += x[i].x * x[i].x + x[i].y * x[i].y + x[i].z * x[i].z + x[i].w * x[i].w;
  }
#pragma unroll
  for (int m = 16; m >= 1; m >>= 1) ss += __shfl_xor(ss, m, 32);

  const float inv = 1.0f / fmaxf(sqrtf(ss), 1e-12f);

  v16bf h, l;
#pragma unroll
  for (int i = 0; i < 4; ++i) {
    const float vals[4] = {x[i].x, x[i].y, x[i].z, x[i].w};
#pragma unroll
    for (int j = 0; j < 4; ++j) {
      const float y  = vals[j] * inv;
      const __bf16 hb = (__bf16)y;
      const float  lo = y - (float)hb;
      h[i * 4 + j] = hb;
      l[i * 4 + j] = (__bf16)lo;
    }
  }
  *(v16bf*)(dhi + lane * 16) = h;
  *(v16bf*)(dlo + lane * 16) = l;
}

// ---------------------------------------------------------------------------
// Kernel 2: bf16x3 split-precision WMMA GEMM + distance epilogue + row sums.
// Block = 256 threads (8 waves). Wave w -> rows [m0, m0+16), 5 n-tiles (80 cols).
// Block tile = 128 (M) x 80 (N).  Grid = (10000/80, 8192/128) = (125, 64).
// ---------------------------------------------------------------------------
__global__ __launch_bounds__(256) void gemm_dist_kernel(
    const __bf16* __restrict__ fn_hi, const __bf16* __restrict__ fn_lo,
    const __bf16* __restrict__ pn_hi, const __bf16* __restrict__ pn_lo,
    const float* __restrict__ dscale,
    float* __restrict__ out, float* __restrict__ rowsum)
{
  const int wave = threadIdx.x >> 5;
  const int lane = threadIdx.x & 31;
  const int lr   = lane & 15;   // row of A-tile / col of B-tile / N in C
  const int half = lane >> 4;   // K-half selector for A/B frags, M-half for C
  const int n0   = blockIdx.x * 80;
  const int m0   = blockIdx.y * 128 + wave * 16;

  // Per-lane fragment base pointers: 16 contiguous bf16 along K.
  const __bf16* aHi = fn_hi + (size_t)(m0 + lr) * DIMK + half * 16;
  const __bf16* aLo = fn_lo + (size_t)(m0 + lr) * DIMK + half * 16;
  const __bf16* bHi = pn_hi + (size_t)(n0 + lr) * DIMK + half * 16;
  const __bf16* bLo = pn_lo + (size_t)(n0 + lr) * DIMK + half * 16;

  v8f acc[5] = {};

  for (int k = 0; k < DIMK; k += 32) {
    const v16bf a_hi = *(const v16bf*)(aHi + k);
    const v16bf a_lo = *(const v16bf*)(aLo + k);
    __builtin_prefetch(aHi + k + 32, 0, 1);
#pragma unroll
    for (int j = 0; j < 5; ++j) {
      const size_t boff = (size_t)j * 16 * DIMK + k;
      const v16bf b_hi = *(const v16bf*)(bHi + boff);
      const v16bf b_lo = *(const v16bf*)(bLo + boff);
      acc[j] = __builtin_amdgcn_wmma_f32_16x16x32_bf16(
          false, a_hi, false, b_hi, (short)0, acc[j], false, false);
      acc[j] = __builtin_amdgcn_wmma_f32_16x16x32_bf16(
          false, a_hi, false, b_lo, (short)0, acc[j], false, false);
      acc[j] = __builtin_amdgcn_wmma_f32_16x16x32_bf16(
          false, a_lo, false, b_hi, (short)0, acc[j], false, false);
    }
  }

  // Epilogue: iso = |s| * sqrt(|1 - cos|); store + per-row partial sums.
  const float s = fabsf(dscale[0]);
  float rp[8];
#pragma unroll
  for (int v = 0; v < 8; ++v) rp[v] = 0.f;

#pragma unroll
  for (int j = 0; j < 5; ++j) {
#pragma unroll
    for (int v = 0; v < 8; ++v) {
      const float iso = s * sqrtf(fabsf(1.0f - acc[j][v]));
      out[(size_t)(m0 + v + 8 * half) * NCOLS + (n0 + j * 16 + lr)] = iso;
      rp[v] += iso;
    }
  }
  // Reduce across the 16-lane N group (xor masks stay within each half-wave).
#pragma unroll
  for (int v = 0; v < 8; ++v) {
#pragma unroll
    for (int m = 1; m < 16; m <<= 1) rp[v] += __shfl_xor(rp[v], m, 32);
  }
  if (lr == 0) {
#pragma unroll
    for (int v = 0; v < 8; ++v)
      atomicAdd(&rowsum[m0 + v + 8 * half], rp[v]);
  }
}

// ---------------------------------------------------------------------------
// Kernel 3: logits = -(iso + rowsum/NCOLS) / T, float4-vectorized in place.
// ---------------------------------------------------------------------------
__global__ __launch_bounds__(256) void finalize_kernel(
    float* __restrict__ out, const float* __restrict__ rowsum,
    const float* __restrict__ temperature, size_t total)
{
  const size_t i4   = (size_t)blockIdx.x * blockDim.x + threadIdx.x;
  const size_t base = i4 * 4;
  if (base >= total) return;

  const float invT = 1.0f / temperature[0];
  const size_t row = base / NCOLS;               // 10000 % 4 == 0: no row split
  const float mean = rowsum[row] * (1.0f / (float)NCOLS);

  float4* p = (float4*)(out + base);
  float4 v = *p;
  v.x = -(v.x + mean) * invT;
  v.y = -(v.y + mean) * invT;
  v.z = -(v.z + mean) * invT;
  v.w = -(v.w + mean) * invT;
  *p = v;
}

// ---------------------------------------------------------------------------
extern "C" void kernel_launch(void* const* d_in, const int* in_sizes, int n_in,
                              void* d_out, int out_size, void* d_ws, size_t ws_size,
                              hipStream_t stream) {
  (void)in_sizes; (void)n_in; (void)out_size; (void)ws_size;

  const float* feats  = (const float*)d_in[0];
  const float* protos = (const float*)d_in[1];
  const float* dscale = (const float*)d_in[2];
  const float* temp   = (const float*)d_in[3];
  float* out = (float*)d_out;

  // Workspace layout (all offsets 32B aligned):
  char* ws = (char*)d_ws;
  const size_t szF = (size_t)NROWS * DIMK * sizeof(__bf16);   // 8,388,608
  const size_t szP = (size_t)NCOLS * DIMK * sizeof(__bf16);   // 10,240,000
  __bf16* fn_hi = (__bf16*)(ws);
  __bf16* fn_lo = (__bf16*)(ws + szF);
  __bf16* pn_hi = (__bf16*)(ws + 2 * szF);
  __bf16* pn_lo = (__bf16*)(ws + 2 * szF + szP);
  float*  rowsum = (float*)(ws + 2 * szF + 2 * szP);

  hipMemsetAsync(rowsum, 0, NROWS * sizeof(float), stream);

  // 18192 rows total, 8 waves per 256-thread block -> 2274 blocks exactly.
  norm_split_kernel<<<2274, 256, 0, stream>>>(feats, protos, fn_hi, fn_lo, pn_hi, pn_lo);

  dim3 g2(NCOLS / 80, NROWS / 128);  // (125, 64)
  gemm_dist_kernel<<<g2, 256, 0, stream>>>(fn_hi, fn_lo, pn_hi, pn_lo, dscale, out, rowsum);

  const size_t total = (size_t)NROWS * NCOLS;
  const int blocks3 = (int)((total / 4 + 255) / 256);  // 80,000
  finalize_kernel<<<blocks3, 256, 0, stream>>>(out, rowsum, temp, total);
}
